// GlobalRepresentationNetwork_41772851921274
// MI455X (gfx1250) — compile-verified, matrix operand order checked by reading, not run on hardware
//
#include <hip/hip_runtime.h>
#include <hip/hip_bf16.h>
#include <math.h>

// ---------------- shapes ----------------
#define BB 4
#define NN 256
#define DD 1024
#define EPSN 1e-5f

typedef __attribute__((ext_vector_type(16))) _Float16 v16h;
typedef __attribute__((ext_vector_type(8)))  _Float16 v8h;
typedef __attribute__((ext_vector_type(8)))  float    v8f;

// WMMA 16x16x32 f16 lane layout (ISA 7.12.2): lane = kg*16 + (m|n);
// halves 0..7 hold K = kg*8 .. kg*8+7 ; halves 8..15 hold K = 16+kg*8 .. 16+kg*8+7.
// With a K-fastest f16 layout each fragment is two aligned 16-byte (b128) loads.
__device__ __forceinline__ v16h load_frag(const _Float16* __restrict__ p, int kg) {
  v8h lo = *(const v8h*)(p + kg * 8);
  v8h hi = *(const v8h*)(p + 16 + kg * 8);
  v16h r;
#pragma unroll
  for (int i = 0; i < 8; i++) { r[i] = lo[i]; r[i + 8] = hi[i]; }
  return r;
}

__device__ __forceinline__ float softf(float v, float t) {
  float s = (v > 0.f) ? 1.f : ((v < 0.f) ? -1.f : 0.f);
  return s * fmaxf(fabsf(v) - t, 0.f);
}

// ---------------- 0) zero the f16 "zero page" (conv SAME padding source) -------
__global__ void zero_f16_kernel(_Float16* __restrict__ p, int n) {
  int i = blockIdx.x * blockDim.x + threadIdx.x;
  if (i < n) p[i] = (_Float16)0.f;
}

// ---------------- 1) brute-force kNN + similarity top-k ------------------------
template <int K>
__global__ void topk_kernel(const float* __restrict__ pts, int* __restrict__ idxout) {
  int gid = blockIdx.x * blockDim.x + threadIdx.x;
  if (gid >= BB * NN) return;
  int b = gid >> 8;
  const float* pb = pts + (size_t)b * NN * 3;
  int i = gid & 255;
  float xi = pb[i * 3 + 0], yi = pb[i * 3 + 1], zi = pb[i * 3 + 2];
  float mx = 0.f, my = 0.f, mz = 0.f;
  for (int j = 0; j < NN; j++) { mx += pb[j * 3]; my += pb[j * 3 + 1]; mz += pb[j * 3 + 2]; }
  mx *= (1.f / NN); my *= (1.f / NN); mz *= (1.f / NN);
  float sqi = xi * xi + yi * yi + zi * zi;
  float cx = xi - mx, cy = yi - my, cz = zi - mz;
  const int KN = K + 1, KS = K - 1;
  float nv[KN]; int ni[KN];
  float sv[KS]; int si[KS];
#pragma unroll
  for (int t = 0; t < KN; t++) { nv[t] = -3.4e38f; ni[t] = 0; }
#pragma unroll
  for (int t = 0; t < KS; t++) { sv[t] = -3.4e38f; si[t] = 0; }
  for (int j = 0; j < NN; j++) {
    float xj = pb[j * 3], yj = pb[j * 3 + 1], zj = pb[j * 3 + 2];
    float dt = xi * xj + yi * yj + zi * zj;
    float sqj = xj * xj + yj * yj + zj * zj;
    float nd = -(sqi + sqj - 2.f * dt);
    if (nd > nv[KN - 1]) {
      int p = KN - 1;
      while (p > 0 && nd > nv[p - 1]) { nv[p] = nv[p - 1]; ni[p] = ni[p - 1]; --p; }
      nv[p] = nd; ni[p] = j;
    }
    float sm = cx * (xj - mx) + cy * (yj - my) + cz * (zj - mz);
    if (sm > sv[KS - 1]) {
      int p = KS - 1;
      while (p > 0 && sm > sv[p - 1]) { sv[p] = sv[p - 1]; si[p] = si[p - 1]; --p; }
      sv[p] = sm; si[p] = j;
    }
  }
  int* op = idxout + (size_t)gid * (2 * K);
#pragma unroll
  for (int t = 0; t < KN; t++) op[t] = ni[t];
#pragma unroll
  for (int t = 0; t < KS; t++) op[KN + t] = si[t];
}

// ---------------- 2) gather + Gram + row sort -> f16 transposed (c-fastest) ----
template <int TK>
__global__ void gram_sort_kernel(const float* __restrict__ pts, const int* __restrict__ idx,
                                 _Float16* __restrict__ srtT) {
  __shared__ float f[TK][3];
  int pid = blockIdx.x;                 // b*NN + n  (n is the conv channel)
  int b = pid >> 8;
  int n = pid & 255;
  int j = threadIdx.x;
  if (j < TK) {
    int id = idx[(size_t)pid * TK + j];
    const float* p = pts + ((size_t)b * NN + id) * 3;
    f[j][0] = p[0]; f[j][1] = p[1]; f[j][2] = p[2];
  }
  __syncthreads();
  if (j < TK) {
    float a0 = f[j][0], a1 = f[j][1], a2 = f[j][2];
    float g[TK];
    for (int l = 0; l < TK; l++) g[l] = a0 * f[l][0] + a1 * f[l][1] + a2 * f[l][2];
    for (int a = 1; a < TK; a++) {           // insertion sort, descending
      float v = g[a]; int p = a;
      while (p > 0 && g[p - 1] < v) { g[p] = g[p - 1]; --p; }
      g[p] = v;
    }
    // layout: srtT[(b*W*W + pos)*256 + n], pos = j*TK + l
    _Float16* op = srtT + ((size_t)b * TK * TK + (size_t)j * TK) * 256 + n;
    for (int l = 0; l < TK; l++) op[(size_t)l * 256] = (_Float16)g[l];
  }
}

// ---------------- 3) pack conv weights: (co,ci,kh,kw) f32 -> [t][co][ci] f16 ---
__global__ void pack_convw_kernel(const float* __restrict__ w, _Float16* __restrict__ wpk) {
  int o = blockIdx.x * blockDim.x + threadIdx.x;
  if (o >= 9 * 256 * 256) return;
  int t = o >> 16;                // kh*3+kw
  int co = (o >> 8) & 255;
  int ci = o & 255;
  wpk[o] = (_Float16)w[((size_t)co * 256 + ci) * 9 + t];
}

// ---------------- 4) plain f32 -> f16 convert (GEMM weights) -------------------
__global__ void cvt_f16_kernel(const float* __restrict__ in, _Float16* __restrict__ out, int n) {
  int i = blockIdx.x * blockDim.x + threadIdx.x;
  if (i < n) out[i] = (_Float16)in[i];
}

// ---------------- 5) conv2d 3x3 SAME 256->256, all-f16 WMMA, 4x cout blocking --
// inT: f16 [b][pos][cin] ; wpk: f16 [t][cout][cin] ; out f32 [b][cout][pos] ;
// outT (optional): f16 [b][pos][cout]. OOB taps read a zeroed f16 page (pointer
// select only -> no per-element masking). One wave computes a 64(cout)x16(pos)
// tile: per k-chunk the B fragment is loaded once and reused by 4 WMMAs.
__global__ void conv2d_wmma_kernel(const _Float16* __restrict__ inT,
                                   const _Float16* __restrict__ wpk,
                                   const float* __restrict__ bias, float* __restrict__ out,
                                   _Float16* __restrict__ outT,
                                   const _Float16* __restrict__ zeroT, int W) {
  int L = threadIdx.x, l15 = L & 15, kg = L >> 4;
  int b = blockIdx.z;
  int pos = blockIdx.x * 16 + l15;          // B-frag column (spatial position)
  int y0 = pos / W, x0 = pos % W;
  int coutBase = blockIdx.y * 64;           // 4 cout tiles per wave
  int WW = W * W;
  v8f acc[4] = {v8f{}, v8f{}, v8f{}, v8f{}};
#pragma unroll
  for (int kh = 0; kh < 3; kh++) {
#pragma unroll
    for (int kw = 0; kw < 3; kw++) {
      int iy = y0 + kh - 1, ix = x0 + kw - 1;
      bool inb = ((unsigned)iy < (unsigned)W) && ((unsigned)ix < (unsigned)W);
      const _Float16* pin =
          inb ? (inT + ((size_t)b * WW + iy * W + ix) * 256) : zeroT;  // ptr select
      const _Float16* pw0 =
          wpk + ((size_t)(kh * 3 + kw) * 256 + coutBase + l15) * 256;
      for (int c0 = 0; c0 < 256; c0 += 32) {
        v16h bf = load_frag(pin + c0, kg);
#pragma unroll
        for (int t = 0; t < 4; t++) {
          v16h a = load_frag(pw0 + (size_t)t * 16 * 256 + c0, kg);
          acc[t] = __builtin_amdgcn_wmma_f32_16x16x32_f16(false, a, false, bf, (short)0,
                                                          acc[t], false, false);
        }
      }
    }
  }
#pragma unroll
  for (int t = 0; t < 4; t++) {
#pragma unroll
    for (int r = 0; r < 8; r++) {
      int co = coutBase + t * 16 + r + kg * 8;
      float v = acc[t][r];
      if (bias) v += bias[co];
      out[((size_t)b * 256 + co) * WW + pos] = v;
      if (outT) outT[((size_t)b * WW + pos) * 256 + co] = (_Float16)v;
    }
  }
}

// ---------------- 6) batchnorm stats / fused apply(+relu)(+f16 transpose) ------
__global__ void bn_stats_kernel(const float* __restrict__ in, float* __restrict__ mean,
                                float* __restrict__ rstd, int C, int S, int Bn) {
  __shared__ float ss[256], sq[256];
  int c = blockIdx.x, t = threadIdx.x;
  float s = 0.f, q = 0.f;
  int cnt = Bn * S;
  for (int i = t; i < cnt; i += 256) {
    int b = i / S, sp = i - b * S;
    float v = in[((size_t)b * C + c) * S + sp];
    s += v; q += v * v;
  }
  ss[t] = s; sq[t] = q;
  __syncthreads();
  for (int o = 128; o > 0; o >>= 1) {
    if (t < o) { ss[t] += ss[t + o]; sq[t] += sq[t + o]; }
    __syncthreads();
  }
  if (t == 0) {
    float m = ss[0] / cnt;
    float v = sq[0] / cnt - m * m;
    mean[c] = m;
    rstd[c] = rsqrtf(v + EPSN);
  }
}

// x layout (B,C,S); optional xT f16 layout [(b*S+sp)]*C + c (K-fastest for WMMA)
__global__ void bn_apply_kernel(float* __restrict__ x, const float* __restrict__ mean,
                                const float* __restrict__ rstd, const float* __restrict__ g,
                                const float* __restrict__ bt, _Float16* __restrict__ xT,
                                int C, int S, int total, int relu) {
  int i = blockIdx.x * blockDim.x + threadIdx.x;
  if (i >= total) return;
  int c = (i / S) % C;
  int b = i / (S * C);
  int sp = i % S;
  float v = (x[i] - mean[c]) * rstd[c] * g[c] + bt[c];
  if (relu) v = fmaxf(v, 0.f);
  x[i] = v;
  if (xT) xT[((size_t)b * S + sp) * C + c] = (_Float16)v;
}

// ---------------- 7) GAP = mean|fm| over spatial -------------------------------
__global__ void gap_kernel(const float* __restrict__ fm, float* __restrict__ gap, int S) {
  __shared__ float ss[128];
  int bc = blockIdx.x, t = threadIdx.x;
  float s = 0.f;
  for (int i = t; i < S; i += 128) s += fabsf(fm[(size_t)bc * S + i]);
  ss[t] = s;
  __syncthreads();
  for (int o = 64; o > 0; o >>= 1) { if (t < o) ss[t] += ss[t + o]; __syncthreads(); }
  if (t == 0) gap[bc] = ss[0] / S;
}

// ---------------- 8) tiny FC 256->128->256, thr = alphas*gap -------------------
__global__ void alphas_kernel(const float* __restrict__ gap, const float* __restrict__ fc1w,
                              const float* __restrict__ fc1b, const float* __restrict__ fc2w,
                              const float* __restrict__ fc2b, float* __restrict__ thr) {
  __shared__ float sg[256], sh[128];
  int b = blockIdx.x, t = threadIdx.x;
  sg[t] = gap[(size_t)b * 256 + t];
  __syncthreads();
  if (t < 128) {
    float s = fc1b[t];
    for (int j = 0; j < 256; j++) s += fc1w[(size_t)t * 256 + j] * sg[j];
    sh[t] = fmaxf(s, 0.f);
  }
  __syncthreads();
  float s = fc2b[t];
  for (int j = 0; j < 128; j++) s += fc2w[(size_t)t * 128 + j] * sh[j];
  thr[(size_t)b * 256 + t] = s * sg[t];
}

// ---------------- 9) soft-threshold + diag combine + global max per point ------
template <int TK>
__global__ void combine_kernel(const float* __restrict__ lf, const float* __restrict__ fm,
                               const float* __restrict__ thr, float* __restrict__ xc, int slot) {
  int pid = blockIdx.x;                 // b*NN + n ; channel == n
  int w = threadIdx.x;
  float th = thr[pid];
  float mx = -3.4e38f;
  if (w < TK) {
    size_t base = (size_t)pid * TK * TK;
    for (int h = 0; h < TK; h++) {
      float lfv = lf[base + h * TK + w];
      float fmv = fm[base + h * TK + w];
      float lfa = softf(fmv, th) + lfv;
      float v = (h == w) ? (softf(lfv, th) + lfv) * lfa : lfa;
      mx = fmaxf(mx, v);
    }
  }
  for (int o = 16; o > 0; o >>= 1) mx = fmaxf(mx, __shfl_xor(mx, o, 32));
  if (w == 0) xc[(size_t)pid * 3 + slot] = mx;
}

// ---------------- 10) conv1d 3->1024 k=3 SAME (bias cancels in BN) -------------
__global__ void mlp1_kernel(const float* __restrict__ xc, const float* __restrict__ w,
                            float* __restrict__ out) {
  int co = blockIdx.x, b = blockIdx.y, n = threadIdx.x;
  float s = 0.f;
#pragma unroll
  for (int t = 0; t < 3; t++) {
    int nn = n + t - 1;
    if (nn >= 0 && nn < NN) {
      const float* xp = xc + ((size_t)b * NN + nn) * 3;
      s += w[((size_t)co * 3 + 0) * 3 + t] * xp[0] +
           w[((size_t)co * 3 + 1) * 3 + t] * xp[1] +
           w[((size_t)co * 3 + 2) * 3 + t] * xp[2];
    }
  }
  out[((size_t)b * DD + co) * NN + n] = s;
}

// ---------------- 11) 1024^3 GEMM, all-f16 WMMA, 4x row blocking ---------------
// actT: f16 [col][k] (col = b*256+n) ; wh: f16 [row][k] ; out f32 (B,1024,256).
// One wave computes a 64(row)x16(col) tile: the activation fragment is loaded
// once per k-chunk and reused by 4 WMMAs (10x b128 -> 4x v_wmma per chunk).
__global__ void gemm1024_wmma_kernel(const _Float16* __restrict__ actT,
                                     const _Float16* __restrict__ wh,
                                     float* __restrict__ out) {
  int L = threadIdx.x, l15 = L & 15, kg = L >> 4;
  int col = blockIdx.x * 16 + l15;
  int b = col >> 8, nn = col & 255;
  int rowBase = blockIdx.y * 64;            // 4 row tiles per wave
  const _Float16* pa0 = wh + (size_t)(rowBase + l15) * DD;
  const _Float16* pb = actT + (size_t)col * DD;
  v8f acc[4] = {v8f{}, v8f{}, v8f{}, v8f{}};
  for (int k0 = 0; k0 < DD; k0 += 32) {
    v16h bf = load_frag(pb + k0, kg);
#pragma unroll
    for (int t = 0; t < 4; t++) {
      v16h a = load_frag(pa0 + (size_t)t * 16 * DD + k0, kg);
      acc[t] = __builtin_amdgcn_wmma_f32_16x16x32_f16(false, a, false, bf, (short)0,
                                                      acc[t], false, false);
    }
  }
#pragma unroll
  for (int t = 0; t < 4; t++) {
#pragma unroll
    for (int r = 0; r < 8; r++) {
      int co = rowBase + t * 16 + r + kg * 8;
      out[((size_t)b * DD + co) * NN + nn] = acc[t][r];
    }
  }
}

// ---------------- 12) out[b,n,c] = a_cf[b,c,n] * fr[b,c,n] ---------------------
__global__ void final_mul_kernel(const float* __restrict__ acf, const float* __restrict__ fr,
                                 float* __restrict__ out) {
  int i = blockIdx.x * blockDim.x + threadIdx.x;
  if (i >= BB * NN * DD) return;
  int c = i % DD;
  int bn = i / DD;
  int n = bn & 255, b = bn >> 8;
  size_t src = ((size_t)b * DD + c) * NN + n;
  out[i] = acf[src] * fr[src];
}

// ---------------- branch driver (host) ----------------
template <int K>
static void run_branch(const float* x, void* const* d_in, int stbase, int slot,
                       int* idxbuf, _Float16* convT_A, _Float16* convT_B, _Float16* wpkc,
                       _Float16* zeroT, float* lf, float* t1, float* t2,
                       float* meanb, float* rstdb, float* gapb, float* thrb, float* xc,
                       hipStream_t stream) {
  constexpr int TK = 2 * K;
  const int W = TK, S = W * W;
  const float* mlp_w = (const float*)d_in[stbase + 0];
  const float* mlp_b = (const float*)d_in[stbase + 1];
  const float* c1_w  = (const float*)d_in[stbase + 2];  // c1_b cancels in BN
  const float* g1    = (const float*)d_in[stbase + 4];
  const float* b1    = (const float*)d_in[stbase + 5];
  const float* c2_w  = (const float*)d_in[stbase + 6];  // c2_b cancels in BN
  const float* g2    = (const float*)d_in[stbase + 8];
  const float* b2    = (const float*)d_in[stbase + 9];
  const float* fc1_w = (const float*)d_in[stbase + 10];
  const float* fc1_b = (const float*)d_in[stbase + 11];
  const float* fc2_w = (const float*)d_in[stbase + 12];
  const float* fc2_b = (const float*)d_in[stbase + 13];

  topk_kernel<K><<<4, 256, 0, stream>>>(x, idxbuf);
  gram_sort_kernel<TK><<<BB * NN, 32, 0, stream>>>(x, idxbuf, convT_A);  // srtT

  const int nw = 9 * 256 * 256;
  dim3 cg(S / 16, 4, BB);                   // 4 cout tiles per wave
  int total = BB * 256 * S;

  // lf = conv(srt, mlp_w) + mlp_b   (f32 for combine, f16T for next conv)
  pack_convw_kernel<<<(nw + 255) / 256, 256, 0, stream>>>(mlp_w, wpkc);
  conv2d_wmma_kernel<<<cg, 32, 0, stream>>>(convT_A, wpkc, mlp_b, lf, convT_B, zeroT, W);
  // t1 = conv(lf, c1_w) ; fm1 = bn(t1) -> f16T in convT_A
  pack_convw_kernel<<<(nw + 255) / 256, 256, 0, stream>>>(c1_w, wpkc);
  conv2d_wmma_kernel<<<cg, 32, 0, stream>>>(convT_B, wpkc, nullptr, t1, nullptr, zeroT, W);
  bn_stats_kernel<<<256, 256, 0, stream>>>(t1, meanb, rstdb, 256, S, BB);
  bn_apply_kernel<<<(total + 255) / 256, 256, 0, stream>>>(t1, meanb, rstdb, g1, b1,
                                                           convT_A, 256, S, total, 0);
  // fm = bn(conv(fm1, c2_w))  (f32 only; consumed by gap/combine)
  pack_convw_kernel<<<(nw + 255) / 256, 256, 0, stream>>>(c2_w, wpkc);
  conv2d_wmma_kernel<<<cg, 32, 0, stream>>>(convT_A, wpkc, nullptr, t2, nullptr, zeroT, W);
  bn_stats_kernel<<<256, 256, 0, stream>>>(t2, meanb, rstdb, 256, S, BB);
  bn_apply_kernel<<<(total + 255) / 256, 256, 0, stream>>>(t2, meanb, rstdb, g2, b2,
                                                           nullptr, 256, S, total, 0);
  gap_kernel<<<BB * 256, 128, 0, stream>>>(t2, gapb, S);
  alphas_kernel<<<BB, 256, 0, stream>>>(gapb, fc1_w, fc1_b, fc2_w, fc2_b, thrb);
  combine_kernel<TK><<<BB * NN, 32, 0, stream>>>(lf, t2, thrb, xc, slot);
}

extern "C" void kernel_launch(void* const* d_in, const int* in_sizes, int n_in,
                              void* d_out, int out_size, void* d_ws, size_t ws_size,
                              hipStream_t stream) {
  (void)in_sizes; (void)n_in; (void)out_size; (void)ws_size;
  const float* x = (const float*)d_in[0];

  // carve workspace (~30 MB), 256B aligned slots
  char* base = (char*)d_ws;
  size_t off = 0;
  auto alloc = [&](size_t bytes) { void* p = base + off; off += (bytes + 255) & ~(size_t)255; return p; };
  int*      idxbuf  = (int*)alloc(1024 * 32 * sizeof(int));
  _Float16* zeroT   = (_Float16*)alloc(256 * 2);                       // zero page
  _Float16* convT_A = (_Float16*)alloc((size_t)BB * 1024 * 256 * 2);   // f16 [b][pos][c]
  _Float16* convT_B = (_Float16*)alloc((size_t)BB * 1024 * 256 * 2);
  _Float16* wpkc    = (_Float16*)alloc((size_t)9 * 256 * 256 * 2);     // packed conv weights
  float*    lf      = (float*)alloc((size_t)BB * 256 * 1024 * 4);
  float*    t1      = (float*)alloc((size_t)BB * 256 * 1024 * 4);
  float*    t2      = (float*)alloc((size_t)BB * 256 * 1024 * 4);
  float*    meanb   = (float*)alloc(1024 * 4);
  float*    rstdb   = (float*)alloc(1024 * 4);
  float*    gapb    = (float*)alloc(BB * 256 * 4);
  float*    thrb    = (float*)alloc(BB * 256 * 4);
  float*    xc      = (float*)alloc((size_t)BB * NN * 3 * 4);
  float*    act0    = (float*)alloc((size_t)BB * DD * NN * 4);
  float*    act1    = (float*)alloc((size_t)BB * DD * NN * 4);
  _Float16* actT    = (_Float16*)alloc((size_t)BB * NN * DD * 2);      // f16 [col][k]
  _Float16* whalf   = (_Float16*)alloc((size_t)DD * DD * 2);           // f16 [row][k]

  zero_f16_kernel<<<1, 256, 0, stream>>>(zeroT, 256);

  // three branches (st1/st2/st3 params at d_in[1..14], [15..28], [29..42])
  run_branch<8 >(x, d_in, 1,  0, idxbuf, convT_A, convT_B, wpkc, zeroT, lf, t1, t2,
                 meanb, rstdb, gapb, thrb, xc, stream);
  run_branch<12>(x, d_in, 15, 1, idxbuf, convT_A, convT_B, wpkc, zeroT, lf, t1, t2,
                 meanb, rstdb, gapb, thrb, xc, stream);
  run_branch<16>(x, d_in, 29, 2, idxbuf, convT_A, convT_B, wpkc, zeroT, lf, t1, t2,
                 meanb, rstdb, gapb, thrb, xc, stream);

  const int tot = BB * DD * NN;

  // mlp1: conv1d 3->1024 (+BN+ReLU; conv bias cancels in BN); emit f16T for GEMM
  mlp1_kernel<<<dim3(DD, BB), 256, 0, stream>>>(xc, (const float*)d_in[43], act0);
  bn_stats_kernel<<<DD, 256, 0, stream>>>(act0, meanb, rstdb, DD, NN, BB);
  bn_apply_kernel<<<(tot + 255) / 256, 256, 0, stream>>>(act0, meanb, rstdb,
      (const float*)d_in[45], (const float*)d_in[46], actT, DD, NN, tot, 1);

  // mlp2..mlp4, ch : 1024^3 WMMA GEMMs + BN + ReLU; each apply refreshes actT
  const int widx[4] = {47, 51, 55, 59};
  float* bufs[2] = {act0, act1};
  for (int s = 0; s < 4; s++) {
    float* gout = bufs[(s + 1) & 1];
    cvt_f16_kernel<<<(DD * DD + 255) / 256, 256, 0, stream>>>((const float*)d_in[widx[s]],
                                                              whalf, DD * DD);
    gemm1024_wmma_kernel<<<dim3(64, 16), 32, 0, stream>>>(actT, whalf, gout);
    bn_stats_kernel<<<DD, 256, 0, stream>>>(gout, meanb, rstdb, DD, NN, BB);
    bn_apply_kernel<<<(tot + 255) / 256, 256, 0, stream>>>(gout, meanb, rstdb,
        (const float*)d_in[widx[s] + 2], (const float*)d_in[widx[s] + 3], actT,
        DD, NN, tot, 1);
  }
  // fr = act1 (after mlp4), a_cf = act0 (after ch)
  final_mul_kernel<<<(tot + 255) / 256, 256, 0, stream>>>(act0, act1, (float*)d_out);
}
